// TransformerBlock_9543417332307
// MI455X (gfx1250) — matrix operand order repaired and checked by reading, not compiled, not run on hardware
//
#include <hip/hip_runtime.h>
#include <hip/hip_bf16.h>
#include <cstdint>

// ---------------- CDNA5 types ----------------
typedef __bf16 bf16_t;
typedef bf16_t v16bf __attribute__((ext_vector_type(16)));
typedef bf16_t v8bf  __attribute__((ext_vector_type(8)));
typedef float  v8f   __attribute__((ext_vector_type(8)));

#define B_   4
#define S_   2048
#define D_   1024
#define H_   16
#define DFF_ 4096
#define DK_  512

// ---------------- async global->LDS (CDNA5, ASYNCcnt) ----------------
__device__ __forceinline__ unsigned lds_off(const void* p) {
  return (unsigned)(size_t)p;  // LDS generic address: low 32 bits = LDS offset
}
__device__ __forceinline__ void async_load_b128(unsigned ldsa, const void* gaddr) {
  asm volatile("global_load_async_to_lds_b128 %0, %1, off" :: "v"(ldsa), "v"(gaddr) : "memory");
}
__device__ __forceinline__ void async_load_b64(unsigned ldsa, const void* gaddr) {
  asm volatile("global_load_async_to_lds_b64 %0, %1, off" :: "v"(ldsa), "v"(gaddr) : "memory");
}
__device__ __forceinline__ void wait_async() {
  asm volatile("s_wait_asynccnt 0" ::: "memory");
}

// ---------------- fragment helpers (ISA 7.12.2 layouts) ----------------
__device__ __forceinline__ v16bf make16(v8bf lo, v8bf hi) {
  v16bf r;
#pragma unroll
  for (int i = 0; i < 8; ++i) { r[i] = lo[i]; r[i + 8] = hi[i]; }
  return r;
}

// A-matrix 16x32 bf16: lane m (0-15) row m, K {0..7,16..23}; lane m+16 row m, K {8..15,24..31}
__device__ __forceinline__ v16bf load_fragA(const bf16_t* base, int row, int stride, int lane) {
  const bf16_t* p = base + (size_t)(row + (lane & 15)) * stride + ((lane >> 4) << 3);
  v8bf lo = *(const v8bf*)(p);
  v8bf hi = *(const v8bf*)(p + 16);
  return make16(lo, hi);
}

// B-matrix 32x16 bf16 from row-major B^T [n][k]: lane n (0-15) K {0..7,16..23}; lane n+16 K {8..15,24..31}
__device__ __forceinline__ v16bf load_fragB(const bf16_t* base, int row, int stride, int lane) {
  const bf16_t* p = base + (size_t)(row + (lane & 15)) * stride + ((lane >> 4) << 3);
  v8bf lo = *(const v8bf*)(p);
  v8bf hi = *(const v8bf*)(p + 16);
  return make16(lo, hi);
}

__device__ __forceinline__ v8f wmma_bf16(v16bf a, v16bf b, v8f c) {
  return __builtin_amdgcn_wmma_f32_16x16x32_bf16(false, a, false, b, (short)0, c, false, false);
}

__device__ __forceinline__ unsigned pack2(bf16_t lo, bf16_t hi) {
  union { bf16_t h[2]; unsigned u; } pk;
  pk.h[0] = lo; pk.h[1] = hi;
  return pk.u;
}

// ---------------- fp32 -> bf16 convert ----------------
__global__ __launch_bounds__(256) void convert_kernel(const float* __restrict__ src,
                                                      bf16_t* __restrict__ dst, int n) {
  int i = (blockIdx.x * 256 + threadIdx.x) * 4;
  if (i + 3 < n) {
    float4 f = *(const float4*)(src + i);
    dst[i + 0] = (bf16_t)f.x; dst[i + 1] = (bf16_t)f.y;
    dst[i + 2] = (bf16_t)f.z; dst[i + 3] = (bf16_t)f.w;
  }
}

// ---------------- bf16 WMMA GEMM: C[M,N] = A[M,K] @ B[K,N] + bias ----------------
// Double-buffered: async A-tile to LDS, B-tile global->regs->transposed LDS store,
// one barrier per k-step, 8 WMMAs per wave per step.
template <bool OUT_BF16>
__global__ __launch_bounds__(256) void gemm_bf16_kernel(const bf16_t* __restrict__ A,
                                                        const bf16_t* __restrict__ B,
                                                        const float* __restrict__ bias,
                                                        void* __restrict__ Cout,
                                                        int M, int N, int K) {
  __shared__ bf16_t Ash[2][128 * 40];   // [m][k], stride 40
  __shared__ bf16_t Bsh[2][128 * 40];   // B^T: [n][k], stride 40
  const int tid  = threadIdx.x;
  const int lane = tid & 31;
  const int wave = tid >> 5;
  const int wm   = wave >> 2;        // 0..1
  const int wn   = wave & 3;         // 0..3
  const int blockM = blockIdx.y * 128;
  const int blockN = blockIdx.x * 128;

  v8f acc[4][2];
#pragma unroll
  for (int mf = 0; mf < 4; ++mf)
#pragma unroll
    for (int nf = 0; nf < 2; ++nf)
#pragma unroll
      for (int r = 0; r < 8; ++r) acc[mf][nf][r] = 0.0f;

  const int arow = tid >> 1, acol = (tid & 1) * 16;   // A staging: 128x32, 32B/thread
  const int bkp  = tid >> 4, bn8  = (tid & 15) * 8;   // B staging: 2 k-rows x 8 n per thread

  auto stageA = [&](int buf, int k0) {
    const bf16_t* src = A + (size_t)(blockM + arow) * K + k0 + acol;
    unsigned off = lds_off(&Ash[buf][arow * 40 + acol]);
    async_load_b128(off, src);
    async_load_b128(off + 16, src + 8);
  };
  auto loadB = [&](int k0, v8bf& b0, v8bf& b1) {
    const bf16_t* src = B + (size_t)(k0 + 2 * bkp) * N + blockN + bn8;
    b0 = *(const v8bf*)(src);
    b1 = *(const v8bf*)(src + N);
  };
  auto storeB = [&](int buf, v8bf b0, v8bf b1) {
#pragma unroll
    for (int i = 0; i < 8; ++i)
      *(unsigned*)&Bsh[buf][(bn8 + i) * 40 + 2 * bkp] = pack2(b0[i], b1[i]);
  };

  const int KT = K >> 5;
  v8bf pb0, pb1;
  stageA(0, 0);
  loadB(0, pb0, pb1);
  storeB(0, pb0, pb1);

  for (int kt = 0; kt < KT; ++kt) {
    wait_async();
    __syncthreads();
    const int cur = kt & 1;
    const bool more = (kt + 1) < KT;
    if (more) {                      // issue next tile while computing this one
      stageA(cur ^ 1, (kt + 1) * 32);
      loadB((kt + 1) * 32, pb0, pb1);
    }

    v16bf af[4], bfr[2];
#pragma unroll
    for (int mf = 0; mf < 4; ++mf) af[mf]  = load_fragA(Ash[cur], wm * 64 + mf * 16, 40, lane);
#pragma unroll
    for (int nf = 0; nf < 2; ++nf) bfr[nf] = load_fragB(Bsh[cur], wn * 32 + nf * 16, 40, lane);
#pragma unroll
    for (int mf = 0; mf < 4; ++mf)
#pragma unroll
      for (int nf = 0; nf < 2; ++nf)
        acc[mf][nf] = wmma_bf16(af[mf], bfr[nf], acc[mf][nf]);

    if (more) storeB(cur ^ 1, pb0, pb1);
  }

  // C/D layout: VGPR r -> rows r / r+8; N = lane&15
#pragma unroll
  for (int mf = 0; mf < 4; ++mf)
#pragma unroll
    for (int nf = 0; nf < 2; ++nf) {
      const int n  = blockN + wn * 32 + nf * 16 + (lane & 15);
      const float bv = bias[n];
      const int rb = blockM + wm * 64 + mf * 16 + ((lane >> 4) * 8);
#pragma unroll
      for (int r = 0; r < 8; ++r) {
        const float val = acc[mf][nf][r] + bv;
        const size_t idx = (size_t)(rb + r) * N + n;
        if constexpr (OUT_BF16) ((bf16_t*)Cout)[idx] = (bf16_t)val;
        else                    ((float*)Cout)[idx]  = val;
      }
    }
}

// ---------------- flash attention: per-wave 16 q-rows, 32-key blocks ----------------
__global__ __launch_bounds__(256) void flash_attn_kernel(const bf16_t* __restrict__ Q,
                                                         const bf16_t* __restrict__ Km,
                                                         const bf16_t* __restrict__ Vm,
                                                         const float* __restrict__ mask,
                                                         bf16_t* __restrict__ Out) {
  __shared__ bf16_t Ksh[2][32 * 40];    // [key][dh]
  __shared__ bf16_t Vt[2][32 * 40];     // V^T: [dh][key]
  __shared__ bf16_t Psh[8][16 * 40];    // per-wave P staging [row][key]

  const int tid = threadIdx.x, lane = tid & 31, wave = tid >> 5;
  const int bh = blockIdx.y;
  const int b  = bh >> 4;               // /H_
  const int h  = bh & 15;
  const int q0 = blockIdx.x * 128 + wave * 16;

  // Q fragment (A-layout) straight from global
  v16bf qf;
  {
    const bf16_t* p = Q + ((size_t)(b * S_ + q0 + (lane & 15))) * DK_ + h * 32 + ((lane >> 4) << 3);
    qf = make16(*(const v8bf*)p, *(const v8bf*)(p + 16));
  }

  float mrow[8], lrow[8];
  v8f o0, o1;
#pragma unroll
  for (int r = 0; r < 8; ++r) { mrow[r] = -1e30f; lrow[r] = 0.0f; o0[r] = 0.0f; o1[r] = 0.0f; }
  const float sc = 0.044194173824159f;  // 1/sqrt(DK)

  const int skey = tid >> 3, sd  = (tid & 7) * 4;   // K staging: 8B/thread, async
  const int vkp  = tid >> 4, vd2 = (tid & 15) * 2;  // V staging: 2 keys x 2 dh per thread

  auto stageK = [&](int buf, int j) {
    const bf16_t* ks = Km + ((size_t)(b * S_ + j + skey)) * DK_ + h * 32 + sd;
    async_load_b64(lds_off(&Ksh[buf][skey * 40 + sd]), ks);
  };
  auto loadV = [&](int j, unsigned& w0, unsigned& w1) {
    const bf16_t* vs = Vm + ((size_t)(b * S_ + j + 2 * vkp)) * DK_ + h * 32 + vd2;
    w0 = *(const unsigned*)(vs);
    w1 = *(const unsigned*)(vs + DK_);
  };
  auto storeV = [&](int buf, unsigned w0, unsigned w1) {  // transpose: Vt[d][key]
    *(unsigned*)&Vt[buf][vd2 * 40 + 2 * vkp]       = (w0 & 0xffffu) | (w1 << 16);
    *(unsigned*)&Vt[buf][(vd2 + 1) * 40 + 2 * vkp] = (w0 >> 16) | (w1 & 0xffff0000u);
  };

  unsigned pv0, pv1;
  stageK(0, 0);
  loadV(0, pv0, pv1);
  storeV(0, pv0, pv1);

  const int JT = S_ / 32;
  for (int jt = 0; jt < JT; ++jt) {
    wait_async();
    __syncthreads();
    const int cur = jt & 1;
    const int j   = jt * 32;
    const bool more = (jt + 1) < JT;
    if (more) {
      stageK(cur ^ 1, j + 32);
      loadV(j + 32, pv0, pv1);
    }

    v16bf kf0 = load_fragB(Ksh[cur], 0, 40, lane);
    v16bf kf1 = load_fragB(Ksh[cur], 16, 40, lane);
    v8f z;
#pragma unroll
    for (int r = 0; r < 8; ++r) z[r] = 0.0f;
    v8f s0 = wmma_bf16(qf, kf0, z);
    v8f s1 = wmma_bf16(qf, kf1, z);

    const int rowoff = (lane >> 4) * 8;
    const int ncol   = lane & 15;
#pragma unroll
    for (int r = 0; r < 8; ++r) {
      const int qr = q0 + r + rowoff;
      const float m0 = mask[(size_t)qr * S_ + j + ncol];
      const float m1 = mask[(size_t)qr * S_ + j + 16 + ncol];
      float a0 = s0[r] * sc - 1e9f * m0;
      float a1 = s1[r] * sc - 1e9f * m1;

      float mx = fmaxf(a0, a1);
#pragma unroll
      for (int off = 1; off < 16; off <<= 1) mx = fmaxf(mx, __shfl_xor(mx, off, 32));
      const float mn    = fmaxf(mrow[r], mx);
      const float alpha = __expf(mrow[r] - mn);
      const float p0 = __expf(a0 - mn);
      const float p1 = __expf(a1 - mn);
      float ps = p0 + p1;
#pragma unroll
      for (int off = 1; off < 16; off <<= 1) ps += __shfl_xor(ps, off, 32);
      lrow[r] = lrow[r] * alpha + ps;
      mrow[r] = mn;
      o0[r] *= alpha;
      o1[r] *= alpha;
      // C-layout -> LDS (A-layout re-read below)
      Psh[wave][(r + rowoff) * 40 + ncol]      = (bf16_t)p0;
      Psh[wave][(r + rowoff) * 40 + 16 + ncol] = (bf16_t)p1;
    }
    asm volatile("s_wait_dscnt 0" ::: "memory"); // wave-local LDS RAW fence

    v16bf pf  = load_fragA(Psh[wave], 0, 40, lane);
    v16bf vf0 = load_fragB(Vt[cur], 0, 40, lane);
    v16bf vf1 = load_fragB(Vt[cur], 16, 40, lane);
    o0 = wmma_bf16(pf, vf0, o0);
    o1 = wmma_bf16(pf, vf1, o1);

    if (more) storeV(cur ^ 1, pv0, pv1);
  }

  const int rowoff = (lane >> 4) * 8;
  const int ncol   = lane & 15;
#pragma unroll
  for (int r = 0; r < 8; ++r) {
    const float inv = 1.0f / lrow[r];
    const int row = q0 + r + rowoff;
    const size_t base = ((size_t)(b * S_ + row)) * DK_ + h * 32;
    Out[base + ncol]      = (bf16_t)(o0[r] * inv);
    Out[base + 16 + ncol] = (bf16_t)(o1[r] * inv);
  }
}

// ---------------- residual (+optional GELU) + LayerNorm over D=1024 ----------------
__device__ __forceinline__ float gelu_tanh_f(float x) {
  const float c = 0.7978845608028654f;
  return 0.5f * x * (1.0f + tanhf(c * (x + 0.044715f * x * x * x)));
}

template <bool GELU_B, bool EMIT_BF16>
__global__ __launch_bounds__(256) void add_ln_kernel(const float* __restrict__ A,
                                                     const float* __restrict__ Bv,
                                                     const float* __restrict__ gamma,
                                                     const float* __restrict__ beta,
                                                     float* __restrict__ outF,
                                                     bf16_t* __restrict__ outB) {
  __shared__ float red[256];
  const int row = blockIdx.x, tid = threadIdx.x;
  const float* a = A  + (size_t)row * D_;
  const float* b = Bv + (size_t)row * D_;
  float v[4];
  float s = 0.0f;
#pragma unroll
  for (int i = 0; i < 4; ++i) {
    const int c = tid + i * 256;
    float t = b[c];
    if (GELU_B) t = gelu_tanh_f(t);
    v[i] = a[c] + t;
    s += v[i];
  }
  red[tid] = s; __syncthreads();
  for (int st = 128; st > 0; st >>= 1) { if (tid < st) red[tid] += red[tid + st]; __syncthreads(); }
  const float mu = red[0] * (1.0f / D_);
  __syncthreads();
  float s2 = 0.0f;
#pragma unroll
  for (int i = 0; i < 4; ++i) { const float d = v[i] - mu; s2 += d * d; }
  red[tid] = s2; __syncthreads();
  for (int st = 128; st > 0; st >>= 1) { if (tid < st) red[tid] += red[tid + st]; __syncthreads(); }
  const float rstd = rsqrtf(red[0] * (1.0f / D_) + 1e-3f);
#pragma unroll
  for (int i = 0; i < 4; ++i) {
    const int c = tid + i * 256;
    const float o = (v[i] - mu) * rstd * gamma[c] + beta[c];
    outF[(size_t)row * D_ + c] = o;
    if (EMIT_BF16) outB[(size_t)row * D_ + c] = (bf16_t)o;
  }
}

// ---------------- host launch ----------------
extern "C" void kernel_launch(void* const* d_in, const int* in_sizes, int n_in,
                              void* d_out, int out_size, void* d_ws, size_t ws_size,
                              hipStream_t stream) {
  (void)in_sizes; (void)n_in; (void)out_size; (void)ws_size;
  const float* x    = (const float*)d_in[0];
  const float* mask = (const float*)d_in[1];
  const float* Wq   = (const float*)d_in[2];
  const float* bq   = (const float*)d_in[3];
  const float* Wk   = (const float*)d_in[4];
  const float* bk   = (const float*)d_in[5];
  const float* Wv   = (const float*)d_in[6];
  const float* bv   = (const float*)d_in[7];
  const float* Wo   = (const float*)d_in[8];
  const float* bo   = (const float*)d_in[9];
  const float* W1   = (const float*)d_in[10];
  const float* b1   = (const float*)d_in[11];
  const float* W2   = (const float*)d_in[12];
  const float* b2   = (const float*)d_in[13];
  const float* g1   = (const float*)d_in[14];
  const float* be1  = (const float*)d_in[15];
  const float* g2   = (const float*)d_in[16];
  const float* be2  = (const float*)d_in[17];

  const size_t R = (size_t)B_ * S_;  // 8192 rows
  char* wsp = (char*)d_ws;
  auto alloc = [&](size_t bytes) { char* p = wsp; wsp += (bytes + 255) & ~(size_t)255; return p; };

  bf16_t* xb   = (bf16_t*)alloc(R * D_ * 2);
  bf16_t* wqb  = (bf16_t*)alloc((size_t)D_ * DK_ * 2);
  bf16_t* wkb  = (bf16_t*)alloc((size_t)D_ * DK_ * 2);
  bf16_t* wvb  = (bf16_t*)alloc((size_t)D_ * DK_ * 2);
  bf16_t* wob  = (bf16_t*)alloc((size_t)DK_ * D_ * 2);
  bf16_t* w1b  = (bf16_t*)alloc((size_t)D_ * DFF_ * 2);
  bf16_t* w2b  = (bf16_t*)alloc((size_t)DFF_ * D_ * 2);
  bf16_t* qb   = (bf16_t*)alloc(R * DK_ * 2);
  bf16_t* kb   = (bf16_t*)alloc(R * DK_ * 2);
  bf16_t* vb   = (bf16_t*)alloc(R * DK_ * 2);
  bf16_t* attnb= (bf16_t*)alloc(R * DK_ * 2);
  float*  proj = (float*)alloc(R * D_ * 4);
  float*  h1   = (float*)alloc(R * D_ * 4);
  bf16_t* h1b  = (bf16_t*)alloc(R * D_ * 2);
  bf16_t* t1b  = (bf16_t*)alloc(R * DFF_ * 2);
  float*  t2   = (float*)alloc(R * D_ * 4);

  // fp32 -> bf16 conversions (inputs only)
  auto conv = [&](const float* s, bf16_t* d, size_t n) {
    convert_kernel<<<dim3((unsigned)(n / 1024)), 256, 0, stream>>>(s, d, (int)n);
  };
  conv(x,  xb,  R * D_);
  conv(Wq, wqb, (size_t)D_ * DK_);
  conv(Wk, wkb, (size_t)D_ * DK_);
  conv(Wv, wvb, (size_t)D_ * DK_);
  conv(Wo, wob, (size_t)DK_ * D_);
  conv(W1, w1b, (size_t)D_ * DFF_);
  conv(W2, w2b, (size_t)DFF_ * D_);

  // QKV projections (bf16 out, consumed by attention WMMA)
  gemm_bf16_kernel<true><<<dim3(DK_ / 128, R / 128), 256, 0, stream>>>(xb, wqb, bq, qb, (int)R, DK_, D_);
  gemm_bf16_kernel<true><<<dim3(DK_ / 128, R / 128), 256, 0, stream>>>(xb, wkb, bk, kb, (int)R, DK_, D_);
  gemm_bf16_kernel<true><<<dim3(DK_ / 128, R / 128), 256, 0, stream>>>(xb, wvb, bv, vb, (int)R, DK_, D_);

  // streaming softmax attention
  flash_attn_kernel<<<dim3(S_ / 128, B_ * H_), 256, 0, stream>>>(qb, kb, vb, mask, attnb);

  // output projection (fp32 out for residual + LN)
  gemm_bf16_kernel<false><<<dim3(D_ / 128, R / 128), 256, 0, stream>>>(attnb, wob, bo, proj, (int)R, D_, DK_);

  // h1 = LN(x + attn_out)  (also emit bf16 for next GEMM)
  add_ln_kernel<false, true><<<dim3((unsigned)R), 256, 0, stream>>>(x, proj, g1, be1, h1, h1b);

  // FFN
  gemm_bf16_kernel<true><<<dim3(DFF_ / 128, R / 128), 256, 0, stream>>>(h1b, w1b, b1, t1b, (int)R, DFF_, D_);
  gemm_bf16_kernel<false><<<dim3(D_ / 128, R / 128), 256, 0, stream>>>(t1b, w2b, b2, t2, (int)R, D_, DFF_);

  // out = LN(h1 + gelu(t2))
  add_ln_kernel<true, false><<<dim3((unsigned)R), 256, 0, stream>>>(h1, t2, g2, be2, (float*)d_out, nullptr);
}
